// KMeansClusterer_23785528885577
// MI455X (gfx1250) — matrix operand order repaired and checked by reading, not compiled, not run on hardware
//
#include <hip/hip_runtime.h>
#include <hip/hip_bf16.h>

// ---------------------------------------------------------------------------
// One k-means iteration on MI455X (gfx1250, wave32).
//   distances  : bf16 WMMA (v_wmma_f32_16x16x32_bf16), f32 accumulate
//   argmin     : skip ||x||^2 (row-constant); wave32 xor-butterfly reduction
//   segment sum: 256 KB per-block LDS accumulator (fits 320 KB WGP LDS),
//                ds_add_f32 atomics, single global atomic flush per block
// ---------------------------------------------------------------------------

typedef __bf16 bf16_t;
typedef __attribute__((ext_vector_type(16))) __bf16 v16bf;
typedef __attribute__((ext_vector_type(8)))  float  v8f;

#define N_PTS   200000
#define DIM     128
#define KC      512
#define NTILES  (KC / 16)        // 32 center tiles
#define DCHUNKS (DIM / 32)       // 4 k-chunks of 32 per WMMA
#define ROWTILES (N_PTS / 16)    // 12500 (divides exactly)
#define MAIN_BLOCKS 128
#define TPB 256

// ---------------------------------------------------------------------------
// ws layout (floats): [gsums 512*128][gcounts 512][c2 512][CB bf16 packed]
// ---------------------------------------------------------------------------

__global__ void kmeans_zero(float* __restrict__ p, int n) {
    int i = blockIdx.x * blockDim.x + threadIdx.x;
    if (i < n) p[i] = 0.0f;
}

// c2[k] = ||centers[k]||^2 ; pack centers into WMMA B-fragments:
// fragment (tile t, chunk dc): 32 lanes x 16 bf16; lane L holds column
// n = L%16, rows k = kb..kb+7, 16+kb..16+kb+7 (kb = L<16 ? 0 : 8) of the 32x16 B chunk.
__global__ void kmeans_prep(const float* __restrict__ centers,
                            float* __restrict__ c2,
                            bf16_t* __restrict__ CB) {
    int gid = blockIdx.x * blockDim.x + threadIdx.x;
    if (gid < KC) {
        const float* c = centers + (size_t)gid * DIM;
        float s = 0.0f;
        #pragma unroll 8
        for (int d = 0; d < DIM; ++d) s += c[d] * c[d];
        c2[gid] = s;
    }
    if (gid < NTILES * DCHUNKS * 32) {
        int lane   = gid & 31;
        int dchunk = (gid >> 5) & (DCHUNKS - 1);
        int tile   = gid >> 7;
        int n      = lane & 15;
        int kb     = (lane < 16) ? 0 : 8;
        const float* src = centers + (size_t)(tile * 16 + n) * DIM + dchunk * 32 + kb;
        bf16_t* dst = CB + (size_t)gid * 16;
        #pragma unroll
        for (int e = 0; e < 16; ++e) dst[e] = (bf16_t)src[(e < 8) ? e : e + 8];
    }
}

__global__ __launch_bounds__(TPB)
void kmeans_main(const float* __restrict__ F,
                 const float* __restrict__ c2,
                 const bf16_t* __restrict__ CB,
                 float* __restrict__ gsums,
                 float* __restrict__ gcounts,
                 int* __restrict__ assign_out) {
    __shared__ float acc[KC * DIM];   // 256 KB private segment-sum accumulator
    __shared__ float cnt[KC];         // 2 KB counts

    const int tid  = threadIdx.x;
    const int lane = tid & 31;
    const int wave = tid >> 5;
    const int n16  = lane & 15;
    const int hi   = lane >> 4;       // 0: lanes 0-15, 1: lanes 16-31

    for (int i = tid; i < KC * DIM; i += TPB) acc[i] = 0.0f;
    for (int i = tid; i < KC; i += TPB)       cnt[i] = 0.0f;
    __syncthreads();

    const v16bf* CBv = (const v16bf*)CB;

    for (int tile = blockIdx.x * 8 + wave; tile < ROWTILES; tile += gridDim.x * 8) {
        const int rowbase = tile * 16;

        // --- build A fragments (16x32 bf16 per chunk), ISA 7.12.2 layout ---
        // lane L: row m = rowbase + L%16; elements e=0..7 -> k=kb..kb+7,
        // e=8..15 -> k=16+kb..16+kb+7, kb = (L<16)?0:8.
        const float* frow = F + (size_t)(rowbase + n16) * DIM;
        const int kb = hi ? 8 : 0;
        v16bf afrag[DCHUNKS];
        #pragma unroll
        for (int dc = 0; dc < DCHUNKS; ++dc) {
            const float* p = frow + dc * 32 + kb;
            const float4 a0 = *(const float4*)(p + 0);
            const float4 a1 = *(const float4*)(p + 4);
            const float4 a2 = *(const float4*)(p + 16);
            const float4 a3 = *(const float4*)(p + 20);
            afrag[dc][0]  = (bf16_t)a0.x; afrag[dc][1]  = (bf16_t)a0.y;
            afrag[dc][2]  = (bf16_t)a0.z; afrag[dc][3]  = (bf16_t)a0.w;
            afrag[dc][4]  = (bf16_t)a1.x; afrag[dc][5]  = (bf16_t)a1.y;
            afrag[dc][6]  = (bf16_t)a1.z; afrag[dc][7]  = (bf16_t)a1.w;
            afrag[dc][8]  = (bf16_t)a2.x; afrag[dc][9]  = (bf16_t)a2.y;
            afrag[dc][10] = (bf16_t)a2.z; afrag[dc][11] = (bf16_t)a2.w;
            afrag[dc][12] = (bf16_t)a3.x; afrag[dc][13] = (bf16_t)a3.y;
            afrag[dc][14] = (bf16_t)a3.z; afrag[dc][15] = (bf16_t)a3.w;
        }

        // --- sweep 512 centers: 32 tiles x 4 WMMA (K=128 total each) ---
        float bestv[8];
        int   besti[8];
        #pragma unroll
        for (int j = 0; j < 8; ++j) { bestv[j] = 3.4e38f; besti[j] = 0; }

        for (int t = 0; t < NTILES; ++t) {
            v8f dacc = {};
            const v16bf* bp = CBv + (size_t)t * DCHUNKS * 32;
            #pragma unroll
            for (int dc = 0; dc < DCHUNKS; ++dc) {
                v16bf bfrag = bp[dc * 32 + lane];
                dacc = __builtin_amdgcn_wmma_f32_16x16x32_bf16(
                    false, afrag[dc], false, bfrag, (short)0, dacc, false, false);
            }
            // lane's column = center index t*16 + n16; ||x||^2 dropped
            const float c2n = c2[t * 16 + n16];
            #pragma unroll
            for (int j = 0; j < 8; ++j) {
                const float s = c2n - 2.0f * dacc[j];
                if (s < bestv[j]) { bestv[j] = s; besti[j] = t * 16 + n16; }
            }
        }

        // --- argmin across the 16 lanes sharing each row (xor butterfly) ---
        #pragma unroll
        for (int j = 0; j < 8; ++j) {
            #pragma unroll
            for (int mask = 8; mask >= 1; mask >>= 1) {
                const float ov = __shfl_xor(bestv[j], mask, 32);
                const int   oi = __shfl_xor(besti[j], mask, 32);
                if (ov < bestv[j] || (ov == bestv[j] && oi < besti[j])) {
                    bestv[j] = ov; besti[j] = oi;
                }
            }
        }
        // lanes 0-15 now all hold winners for rows m=0..7 (besti[j] -> m=j);
        // lanes 16-31 for rows m=8..15.
        if (lane == 0) {
            #pragma unroll
            for (int j = 0; j < 8; ++j) assign_out[rowbase + j] = besti[j];
        }
        if (lane == 16) {
            #pragma unroll
            for (int j = 0; j < 8; ++j) assign_out[rowbase + 8 + j] = besti[j];
        }

        // --- segment-sum this wave's 16 rows into the LDS accumulator ---
        #pragma unroll
        for (int r = 0; r < 16; ++r) {
            const int idx = __shfl(besti[r & 7], (r < 8) ? 0 : 16, 32);
            const float4 v = *(const float4*)(F + (size_t)(rowbase + r) * DIM + lane * 4);
            float* ap = acc + (size_t)idx * DIM + lane * 4;
            atomicAdd(ap + 0, v.x);
            atomicAdd(ap + 1, v.y);
            atomicAdd(ap + 2, v.z);
            atomicAdd(ap + 3, v.w);
            if (lane == 0) atomicAdd(&cnt[idx], 1.0f);
        }
    }

    // --- one global flush per block ---
    __syncthreads();
    for (int i = tid; i < KC * DIM; i += TPB) {
        const float v = acc[i];
        if (v != 0.0f) atomicAdd(&gsums[i], v);
    }
    for (int i = tid; i < KC; i += TPB) {
        const float v = cnt[i];
        if (v != 0.0f) atomicAdd(&gcounts[i], v);
    }
}

__global__ void kmeans_final(const float* __restrict__ gsums,
                             const float* __restrict__ gcounts,
                             float* __restrict__ out) {
    int i = blockIdx.x * blockDim.x + threadIdx.x;
    if (i < KC * DIM) {
        const float c = gcounts[i >> 7];   // i / DIM
        out[i] = gsums[i] / fmaxf(c, 1.0f);
    }
}

extern "C" void kernel_launch(void* const* d_in, const int* in_sizes, int n_in,
                              void* d_out, int out_size, void* d_ws, size_t ws_size,
                              hipStream_t stream) {
    const float* F       = (const float*)d_in[0];   // [N, D] f32
    const float* centers = (const float*)d_in[1];   // [K, D] f32

    float* ws_f    = (float*)d_ws;
    float* gsums   = ws_f;                          // KC*DIM
    float* gcounts = ws_f + KC * DIM;               // KC
    float* c2      = ws_f + KC * DIM + KC;          // KC
    bf16_t* CB     = (bf16_t*)(ws_f + KC * DIM + 2 * KC);  // packed B frags

    float* out_centers = (float*)d_out;                     // [K, D]
    int*   assign_out  = (int*)((float*)d_out + KC * DIM);  // [N] int32

    const int nzero = KC * DIM + KC;
    kmeans_zero<<<(nzero + TPB - 1) / TPB, TPB, 0, stream>>>(ws_f, nzero);
    kmeans_prep<<<(NTILES * DCHUNKS * 32 + TPB - 1) / TPB, TPB, 0, stream>>>(centers, c2, CB);
    kmeans_main<<<MAIN_BLOCKS, TPB, 0, stream>>>(F, c2, CB, gsums, gcounts, assign_out);
    kmeans_final<<<(KC * DIM + TPB - 1) / TPB, TPB, 0, stream>>>(gsums, gcounts, out_centers);
}
